// CSpace_68822555951616
// MI455X (gfx1250) — matrix operand order repaired and checked by reading, not compiled, not run on hardware
//
#include <hip/hip_runtime.h>

// ============================================================================
// Resonator bank = geometric-kernel convolution = first-order complex IIR.
//   ker[c,t] = g_c * eig_c^t  (g = kernels_re[c,0], eig = kernels[c,1]/g)
//   forward : y[t] = eig*y[t-1] + g*x[t]        (truncation term ~1e-11, dropped)
//   backward: same recurrence on time-reversed x, output re-reversed.
//
// MI455X roofline: output = 393 MB -> ~17us store floor @ 23.3 TB/s (stores are
// issued non-temporal so the 192MB L2 keeps audio + carries instead). Inner
// work per 32-sample block is a lower-triangular Toeplitz matmul done with
// v_wmma_f32_16x16x32_f16 (f32 accum); exact f32 carries come from a wave32
// parallel affine scan so f16 error does not compound across blocks.
// ============================================================================

typedef __attribute__((ext_vector_type(16))) _Float16 v16h;
typedef __attribute__((ext_vector_type(8)))  float    v8f;

#define SR_T   48000
#define KLEN   24000
#define NCH    64
#define NBAT   8
#define BLK    32
#define NBLKS  1500            // 48000 / 32
#define NCOLS  (NBLKS * NBAT)  // 12000 (batch,block) columns per direction
#define NTILES (NCOLS / 16)    // 750 column-tiles per (dir, channel)
#define TPW    10              // tiles per wave in phase 2 (B-frag amortization)
#define WPDC   (NTILES / TPW)  // 75 waves per (dir, channel)
#define PW_STRIDE 34           // per-channel: eig^0..eig^32 (float2) + (g,0)
#define CARR_OFF  32768        // byte offset of carry array in d_ws

// ---------------------------------------------------------------------------
// Phase 0: per-channel constants: powers eig^0..32 and gain g.
// ---------------------------------------------------------------------------
__global__ void resonator_init(const float* __restrict__ kre,
                               const float* __restrict__ kim,
                               float2* __restrict__ pwbuf) {
  int c = threadIdx.x;
  if (c >= NCH) return;
  float g  = kre[(size_t)c * KLEN + 0];        // ker[c,0] = g (real, eig^0=1)
  float er = kre[(size_t)c * KLEN + 1] / g;    // eig = ker[c,1] / g
  float ei = kim[(size_t)c * KLEN + 1] / g;
  float pr = 1.f, pi = 0.f;
  for (int j = 0; j <= 32; ++j) {
    pwbuf[c * PW_STRIDE + j] = make_float2(pr, pi);
    float nr = pr * er - pi * ei;
    float ni = pr * ei + pi * er;
    pr = nr; pi = ni;
  }
  pwbuf[c * PW_STRIDE + 33] = make_float2(g, 0.f);
}

// ---------------------------------------------------------------------------
// Phase 1: carries at every 32-sample block boundary, exact f32.
// One wave per sequence (dir,b,c) = 1024 waves. Each lane owns 47 blocks,
// composes its chunk as a complex affine (A,S): state' = A*state + S, then a
// 5-step wave32 Hillis-Steele scan (__shfl_up) stitches lane chunks together.
// ---------------------------------------------------------------------------
__global__ void carry_scan(const float* __restrict__ audio,
                           const float2* __restrict__ pwbuf,
                           float2* __restrict__ carr) {
  int w    = blockIdx.x * 8 + (threadIdx.x >> 5);   // 0..1023
  int lane = threadIdx.x & 31;
  int dir  = w >> 9;
  int b    = (w >> 6) & 7;
  int c    = w & 63;

  float2 eig = pwbuf[c * PW_STRIDE + 1];
  float2 e32 = pwbuf[c * PW_STRIDE + 32];
  float  g   = pwbuf[c * PW_STRIDE + 33].x;

  const int CH = 47;                       // 47*32 lanes >= 1500 blocks
  int b0 = lane * CH;
  int b1 = min(b0 + CH, NBLKS);
  const float* xb = audio + (size_t)b * SR_T;
  int off = dir ? (SR_T - 1) : 0;          // sample index = off + sgn*ts
  int sgn = dir ? -1 : 1;

  // Pass A: chunk-local affine (A,S), S from zero state, A = eig^(32*nblks).
  float ar = 1.f, ai = 0.f, sr = 0.f, si = 0.f;
  for (int blk = b0; blk < b1; ++blk) {
    int t0 = blk * BLK;
#pragma unroll
    for (int j = 0; j < BLK; ++j) {
      int ts = t0 + j;
      float x = xb[off + sgn * ts];
      float nr = eig.x * sr - eig.y * si + g * x;
      float ni = eig.x * si + eig.y * sr;
      sr = nr; si = ni;
    }
    float tr = ar * e32.x - ai * e32.y;
    float ti = ar * e32.y + ai * e32.x;
    ar = tr; ai = ti;
  }

  // Inclusive scan of affine compose: (Ap,Sp) then (Ac,Sc) -> (Ac*Ap, Ac*Sp+Sc)
  for (int o = 1; o < 32; o <<= 1) {
    float par = __shfl_up(ar, o, 32);
    float pai = __shfl_up(ai, o, 32);
    float psr = __shfl_up(sr, o, 32);
    float psi = __shfl_up(si, o, 32);
    if (lane >= o) {
      float nsr = ar * psr - ai * psi + sr;
      float nsi = ar * psi + ai * psr + si;
      float nar = ar * par - ai * pai;
      float nai = ar * pai + ai * par;
      sr = nsr; si = nsi; ar = nar; ai = nai;
    }
  }
  // Exclusive prefix = state entering this lane's chunk.
  float xsr = __shfl_up(sr, 1, 32);
  float xsi = __shfl_up(si, 1, 32);
  if (lane == 0) { xsr = 0.f; xsi = 0.f; }

  // Pass B: emit carry at each block start, then advance state.
  sr = xsr; si = xsi;
  float2* cc = carr + (size_t)(dir * NCH + c) * NCOLS;
  for (int blk = b0; blk < b1; ++blk) {
    cc[blk * 8 + b] = make_float2(sr, si);   // q = blk*8 + b
    int t0 = blk * BLK;
#pragma unroll
    for (int j = 0; j < BLK; ++j) {
      int ts = t0 + j;
      float x = xb[off + sgn * ts];
      float nr = eig.x * sr - eig.y * si + g * x;
      float ni = eig.x * si + eig.y * sr;
      sr = nr; si = ni;
    }
  }
}

// ---------------------------------------------------------------------------
// Phase 2: WMMA block materialization. One wave per (dir, c, group of TPW
// 16-column tiles) -- the per-channel Toeplitz B fragments are built once and
// reused across TPW tiles.
//   D(16 cols x 16 times) = A(16 cols x 32 samples) * B(32 x 16 Toeplitz)
// Two D tiles cover times 0..15 (B1, lower-tri) and 16..31 (B2, full K=32),
// times re/im -> 4 x v_wmma_f32_16x16x32_f16 per tile. Epilogue adds
// eig^{t+1}*carry and stores non-temporal, coalesced (consecutive lanes ->
// consecutive output times).
// ---------------------------------------------------------------------------
__global__ void __launch_bounds__(256)
wmma_blocks(const float* __restrict__ audio,
            const float2* __restrict__ pwbuf,
            const float2* __restrict__ carr,
            float* __restrict__ out) {
  int w    = blockIdx.x * 8 + (threadIdx.x >> 5);  // 0..9599
  int lane = threadIdx.x & 31;
  int dir  = w / (NCH * WPDC);
  int rem  = w - dir * (NCH * WPDC);
  int c    = rem / WPDC;
  int widx = rem - c * WPDC;
  int n  = lane & 15;   // time-within-halfblock (D's N), or column (A's M)
  int hi = lane >> 4;
  int off = dir ? (SR_T - 1) : 0;
  int sgn = dir ? -1 : 1;

  const float2* pw = pwbuf + c * PW_STRIDE;
  float g = pw[33].x;

  // B fragments: B[k][n], lane holds rows k = 16*hi + idx, column n.
  //   B1[k][n] = g*eig^(n-k)    for k <= n    (times 0..15)
  //   B2[k][n] = g*eig^(16+n-k) for k <= 16+n (times 16..31)
  v16h b1re = {}, b1im = {}, b2re = {}, b2im = {};
#pragma unroll
  for (int idx = 0; idx < 16; ++idx) {
    int k = hi * 16 + idx;
    int e1 = n - k;
    if (e1 >= 0) {
      float2 p = pw[e1];
      b1re[idx] = (_Float16)(g * p.x);
      b1im[idx] = (_Float16)(g * p.y);
    }
    int e2 = 16 + n - k;
    if (e2 >= 0) {
      float2 p = pw[e2];
      b2re[idx] = (_Float16)(g * p.x);
      b2im[idx] = (_Float16)(g * p.y);
    }
  }

  // Hoisted per-wave epilogue constants.
  float2 cf1 = pw[n + 1];        // times 0..15  -> eig^(n+1)
  float2 cf2 = pw[n + 17];       // times 16..31 -> eig^(n+17)
  const float2* cc = carr + (size_t)(dir * NCH + c) * NCOLS;
  int kb = hi * 16;              // A-fragment K base for this lane half

  for (int i = 0; i < TPW; ++i) {
    int tile  = widx * TPW + i;
    int qbase = tile * 16;

    // A fragment: column m = n -> (batch, block); rows k = samples in block.
    // Lane needs k in [kb, kb+16): vector loads, dir-uniform.
    int q   = qbase + n;
    int bb  = q & 7;
    int blk = q >> 3;
    int t0  = blk * BLK;
    const float* xb = audio + (size_t)bb * SR_T;
    v16h ax;
    if (dir == 0) {
      const float4 a0 = *(const float4*)(xb + t0 + kb);
      const float4 a1 = *(const float4*)(xb + t0 + kb + 4);
      const float4 a2 = *(const float4*)(xb + t0 + kb + 8);
      const float4 a3 = *(const float4*)(xb + t0 + kb + 12);
      ax[0]=(_Float16)a0.x; ax[1]=(_Float16)a0.y; ax[2]=(_Float16)a0.z; ax[3]=(_Float16)a0.w;
      ax[4]=(_Float16)a1.x; ax[5]=(_Float16)a1.y; ax[6]=(_Float16)a1.z; ax[7]=(_Float16)a1.w;
      ax[8]=(_Float16)a2.x; ax[9]=(_Float16)a2.y; ax[10]=(_Float16)a2.z; ax[11]=(_Float16)a2.w;
      ax[12]=(_Float16)a3.x; ax[13]=(_Float16)a3.y; ax[14]=(_Float16)a3.z; ax[15]=(_Float16)a3.w;
    } else {
      // sample(k) = xb[SR_T-1-t0-k]: descending contiguous runs, 16B aligned.
      const float* p = xb + (SR_T - 4 - t0 - kb);
      const float4 a0 = *(const float4*)(p);        // k = kb+3 .. kb   (rev)
      const float4 a1 = *(const float4*)(p - 4);    // k = kb+7 .. kb+4 (rev)
      const float4 a2 = *(const float4*)(p - 8);    // k = kb+11.. kb+8 (rev)
      const float4 a3 = *(const float4*)(p - 12);   // k = kb+15.. kb+12(rev)
      ax[0]=(_Float16)a0.w; ax[1]=(_Float16)a0.z; ax[2]=(_Float16)a0.y; ax[3]=(_Float16)a0.x;
      ax[4]=(_Float16)a1.w; ax[5]=(_Float16)a1.z; ax[6]=(_Float16)a1.y; ax[7]=(_Float16)a1.x;
      ax[8]=(_Float16)a2.w; ax[9]=(_Float16)a2.z; ax[10]=(_Float16)a2.y; ax[11]=(_Float16)a2.x;
      ax[12]=(_Float16)a3.w; ax[13]=(_Float16)a3.z; ax[14]=(_Float16)a3.y; ax[15]=(_Float16)a3.x;
    }

    v8f zero = {};
    v8f d1re = __builtin_amdgcn_wmma_f32_16x16x32_f16(false, ax, false, b1re, (short)0, zero, false, false);
    v8f d1im = __builtin_amdgcn_wmma_f32_16x16x32_f16(false, ax, false, b1im, (short)0, zero, false, false);
    v8f d2re = __builtin_amdgcn_wmma_f32_16x16x32_f16(false, ax, false, b2re, (short)0, zero, false, false);
    v8f d2im = __builtin_amdgcn_wmma_f32_16x16x32_f16(false, ax, false, b2im, (short)0, zero, false, false);

    // Epilogue: y[local] += eig^(local+1) * carry;  D lane holds time N = n,
    // columns M = r + 8*hi across the 8 accumulator VGPRs. Fully unrolled so
    // accumulator indexing is static (no v_movrels).
#pragma unroll
    for (int r = 0; r < 8; ++r) {
      int m    = r + 8 * hi;
      int qq   = qbase + m;
      int ob   = qq & 7;
      int oblk = qq >> 3;
      float2 s = cc[qq];
      float y1r = d1re[r] + cf1.x * s.x - cf1.y * s.y;
      float y1i = d1im[r] + cf1.x * s.y + cf1.y * s.x;
      float y2r = d2re[r] + cf2.x * s.x - cf2.y * s.y;
      float y2i = d2im[r] + cf2.x * s.y + cf2.y * s.x;
      int t1 = oblk * BLK + n;
      int o1 = off + sgn * t1;
      int o2 = off + sgn * (t1 + 16);
      // planes: [RealF(0..63) ImagF(64..127) RealB(128..191) ImagB(192..255)]
      float* base = out + ((size_t)ob * 256 + (size_t)dir * 128 + c) * SR_T;
      __builtin_nontemporal_store(y1r, base + o1);
      __builtin_nontemporal_store(y1i, base + (size_t)64 * SR_T + o1);
      __builtin_nontemporal_store(y2r, base + o2);
      __builtin_nontemporal_store(y2i, base + (size_t)64 * SR_T + o2);
    }
  }
}

// ---------------------------------------------------------------------------
extern "C" void kernel_launch(void* const* d_in, const int* in_sizes, int n_in,
                              void* d_out, int out_size, void* d_ws, size_t ws_size,
                              hipStream_t stream) {
  const float* audio = (const float*)d_in[0];
  const float* kre   = (const float*)d_in[1];
  const float* kim   = (const float*)d_in[2];
  float* out = (float*)d_out;

  // d_ws: [0, 32KB) eig-power table; [32KB, 32KB+12.3MB) block carries.
  float2* pwbuf = (float2*)d_ws;
  float2* carr  = (float2*)((char*)d_ws + CARR_OFF);

  resonator_init<<<1, 64, 0, stream>>>(kre, kim, pwbuf);
  carry_scan<<<128, 256, 0, stream>>>(audio, pwbuf, carr);          // 1024 waves
  wmma_blocks<<<(2 * NCH * WPDC) / 8, 256, 0, stream>>>(audio, pwbuf, carr, out); // 9600 waves
}
